// LinformerSeqEncoder_32392643346491
// MI455X (gfx1250) — matrix operand order repaired and checked by reading, not compiled, not run on hardware
//
#include <hip/hip_runtime.h>
#include <hip/hip_bf16.h>
#include <math.h>

#define BB 8
#define NN 10000
#define DD 128
#define HH 4
#define DHH 32
#define KK 256
#define MM 5000
#define RTOT (BB * NN)

typedef __attribute__((ext_vector_type(16))) _Float16 v16h;
typedef __attribute__((ext_vector_type(8)))  float    v8f;

// CDNA5 async global->LDS path (ASYNCcnt-tracked), guarded so either toolchain compiles.
// Probe result: builtin takes (int4-vector*, int4-vector*, imm offset, imm cpol).
#if defined(__gfx1250__) && __has_builtin(__builtin_amdgcn_global_load_async_to_lds_b128) && \
    __has_builtin(__builtin_amdgcn_s_wait_asynccnt)
#define HAS_ASYNC_LDS 1
typedef int v4ia __attribute__((ext_vector_type(4)));
#define ASYNC_COPY_B128(gsrc, ldst) \
  __builtin_amdgcn_global_load_async_to_lds_b128((v4ia*)(gsrc), (v4ia*)(ldst), 0, 0)
#define ASYNC_WAIT() __builtin_amdgcn_s_wait_asynccnt(0)
#else
#define HAS_ASYNC_LDS 0
#endif

// ---------------------------------------------------------------------------
// WMMA helpers (CDNA5 wave32 layouts, CDNA5 ISA §7.12.2)
// ---------------------------------------------------------------------------
static __device__ __forceinline__ v8f wmma_f16(v16h a, v16h b, v8f c) {
  return __builtin_amdgcn_wmma_f32_16x16x32_f16(
      /*neg_a=*/false, a, /*neg_b=*/false, b,
      /*c_mod=*/(short)0, c, /*reuse_a=*/false, /*reuse_b=*/false);
}

// A-fragment: 16x32 f16 tile, source row-major [rows][ld] (rows = M, cols = K)
static __device__ __forceinline__ v16h load_frag_a(const _Float16* p, int ld) {
  const int lane = threadIdx.x & 31;
  const int m = lane & 15, sel = lane >> 4;
  v16h a;
#pragma unroll
  for (int e = 0; e < 16; ++e) {
    int k = (sel << 4) + e;
    a[e] = p[m * ld + k];
  }
  return a;
}

// B-fragment from "weight style" layout: row-major [outCols][K]; p -> [col0][k0]
static __device__ __forceinline__ v16h load_frag_b_nk(const _Float16* p, int ld) {
  const int lane = threadIdx.x & 31;
  const int n = lane & 15, sel = lane >> 4;
  v16h b;
#pragma unroll
  for (int e = 0; e < 16; ++e) b[e] = p[n * ld + (sel << 4) + e];
  return b;
}

// B-fragment from contraction-major layout: row-major [K][outCols]; p -> [k0][col0]
static __device__ __forceinline__ v16h load_frag_b_kn(const _Float16* p, int ld) {
  const int lane = threadIdx.x & 31;
  const int n = lane & 15, sel = lane >> 4;
  v16h b;
#pragma unroll
  for (int e = 0; e < 16; ++e) b[e] = p[((sel << 4) + e) * ld + n];
  return b;
}

// ---------------------------------------------------------------------------
// f32 -> f16 conversions (plain + transposing, for proj_k/proj_v -> [K,N])
// ---------------------------------------------------------------------------
__global__ void k_f32_to_f16(const float* __restrict__ s, _Float16* __restrict__ d, int n) {
  int i = blockIdx.x * blockDim.x + threadIdx.x;
  if (i < n) d[i] = (_Float16)s[i];
}

__global__ void k_cvt_transpose(const float* __restrict__ s, _Float16* __restrict__ d,
                                int rows, int cols) {  // src [rows][cols] -> dst [cols][rows]
  int i = blockIdx.x * blockDim.x + threadIdx.x;
  if (i >= rows * cols) return;
  int r = i / cols, c = i - r * cols;
  d[(size_t)c * rows + r] = (_Float16)s[i];
}

// ---------------------------------------------------------------------------
// Embedding + sinusoidal positional encoding
// ---------------------------------------------------------------------------
__global__ void k_embed(const float* __restrict__ x, const float* __restrict__ ew,
                        const float* __restrict__ eb, float* __restrict__ xe) {
  int idx = blockIdx.x * blockDim.x + threadIdx.x;
  if (idx >= RTOT * DD) return;
  int d = idx & (DD - 1);
  int r = idx >> 7;  // b*N + n
  int n = r % NN;
  float freq = __expf(-(float)(d & ~1) * (logf(10000.0f) / (float)DD));
  float ang = (float)n * freq;
  float pe = (d & 1) ? __cosf(ang) : __sinf(ang);
  xe[idx] = x[r] * ew[d] + eb[d] + pe;
}

// ---------------------------------------------------------------------------
// LayerNorm over D=128, one wave32 per row, f16 output
// ---------------------------------------------------------------------------
__global__ void k_layernorm(const float* __restrict__ xe, const float* __restrict__ g,
                            const float* __restrict__ bta, _Float16* __restrict__ out) {
  int row = blockIdx.x * 8 + (threadIdx.x >> 5);
  int lane = threadIdx.x & 31;
  float4 xv = *(const float4*)&xe[(size_t)row * DD + lane * 4];
  float sm = xv.x + xv.y + xv.z + xv.w;
#pragma unroll
  for (int m = 1; m < 32; m <<= 1) sm += __shfl_xor(sm, m);
  float mu = sm * (1.0f / DD);
  float d0 = xv.x - mu, d1 = xv.y - mu, d2 = xv.z - mu, d3 = xv.w - mu;
  float sq = d0 * d0 + d1 * d1 + d2 * d2 + d3 * d3;
#pragma unroll
  for (int m = 1; m < 32; m <<= 1) sq += __shfl_xor(sq, m);
  float rstd = rsqrtf(sq * (1.0f / DD) + 1e-5f);
  int c = lane * 4;
  size_t base = (size_t)row * DD + c;
  out[base + 0] = (_Float16)(d0 * rstd * g[c + 0] + bta[c + 0]);
  out[base + 1] = (_Float16)(d1 * rstd * g[c + 1] + bta[c + 1]);
  out[base + 2] = (_Float16)(d2 * rstd * g[c + 2] + bta[c + 2]);
  out[base + 3] = (_Float16)(d3 * rstd * g[c + 3] + bta[c + 3]);
}

// ---------------------------------------------------------------------------
// Token GEMM: Y[r,c] = sum_k X[r,k] * W[c,k]   (X f16 [R,128], W f16 [128,128])
// 64 rows / block (8 waves): 4 row-tiles x 8 col-tiles, 4 WMMA tiles per wave.
// Full 32 KB weight panel resident in LDS, amortized over 64 rows.
// ---------------------------------------------------------------------------
__global__ void k_gemm_tok(const _Float16* __restrict__ X, const _Float16* __restrict__ W,
                           const float* __restrict__ bias, _Float16* __restrict__ out16,
                           float* __restrict__ acc32) {
  __shared__ __align__(16) _Float16 sW[DD * DD];
  __shared__ __align__(16) _Float16 sX[64][40];
  const int tid = threadIdx.x;
  const int wave = tid >> 5;
  const int rt = wave >> 1;          // row tile 0..3
  const int cbase = (wave & 1) * 4;  // col tile group
  const int r0 = blockIdx.x * 64;

#if HAS_ASYNC_LDS
#pragma unroll
  for (int i = 0; i < 8; ++i) {  // 2048 x 16B = 32 KB panel
    int u = tid + i * 256;
    ASYNC_COPY_B128((const uint4*)W + u, (uint4*)sW + u);
  }
#else
  {
    const uint4* wsrc = (const uint4*)W;
    uint4* wdst = (uint4*)sW;
    for (int i = tid; i < DD * DD / 8; i += 256) wdst[i] = wsrc[i];
  }
#endif

  v8f acc[4] = {};
  for (int kt = 0; kt < DD / 32; ++kt) {
    int k0 = kt * 32;
    {  // stage X tile [64][32]
      int rr = tid >> 2, grp = tid & 3;
#if HAS_ASYNC_LDS
      ASYNC_COPY_B128(&X[(size_t)(r0 + rr) * DD + k0 + grp * 8], &sX[rr][grp * 8]);
      ASYNC_WAIT();
#else
      *(uint4*)&sX[rr][grp * 8] =
          *(const uint4*)&X[(size_t)(r0 + rr) * DD + k0 + grp * 8];
#endif
    }
    __syncthreads();
    v16h a = load_frag_a(&sX[rt * 16][0], 40);
#pragma unroll
    for (int t = 0; t < 4; ++t) {
      int col0 = (cbase + t) * 16;
      v16h b = load_frag_b_nk(&sW[col0 * DD + k0], DD);
      acc[t] = wmma_f16(a, b, acc[t]);
    }
    __syncthreads();
  }

  const int lane = tid & 31, sel = lane >> 4, cn = lane & 15;
#pragma unroll
  for (int t = 0; t < 4; ++t) {
    int col = (cbase + t) * 16 + cn;
    float bv = bias ? bias[col] : 0.0f;
#pragma unroll
    for (int e = 0; e < 8; ++e) {
      size_t off = (size_t)(r0 + rt * 16 + e + sel * 8) * DD + col;
      float v = acc[t][e] + bv;
      if (out16) out16[off] = (_Float16)v;
      if (acc32) acc32[off] += v;
    }
  }
}

// ---------------------------------------------------------------------------
// Linformer projection: outkv[b,kk,d] = sum_n KV[b,n,d] * PT[kk,n]
// PT is pre-transposed [K,N] f16 so A-fragments are contiguous vector loads.
// B = KV tile staged in LDS (async path on full steps); contraction over N.
// ---------------------------------------------------------------------------
__global__ void k_proj(const _Float16* __restrict__ KV, const _Float16* __restrict__ PT,
                       _Float16* __restrict__ outkv) {
  __shared__ __align__(16) _Float16 sB[32][136];
  const int tid = threadIdx.x, wave = tid >> 5, lane = tid & 31;
  const int b = blockIdx.y, kk0 = blockIdx.x * 16;
  const int m = lane & 15, sel = lane >> 4;
  v8f acc[2] = {};
  const int steps = (NN + 31) / 32;  // 313, last step has 16 valid rows
  for (int st = 0; st < steps; ++st) {
    int n0 = st * 32;
    if (n0 + 32 <= NN) {
#if HAS_ASYNC_LDS
#pragma unroll
      for (int i = 0; i < 4; ++i) {  // 512 x 16B = 8 KB tile
        int u = tid + i * 128;
        int rr = u >> 3, cc = (u & 7) * 8;
        ASYNC_COPY_B128(&KV[(size_t)(b * NN + n0 + rr) * DD + cc], &sB[rr][cc]);
      }
      ASYNC_WAIT();
#else
#pragma unroll
      for (int i = 0; i < 4; ++i) {
        int u = tid + i * 128;
        int rr = u >> 3, cc = (u & 7) * 8;
        *(uint4*)&sB[rr][cc] = *(const uint4*)&KV[(size_t)(b * NN + n0 + rr) * DD + cc];
      }
#endif
    } else {  // tail: zero-padded manual staging
#pragma unroll
      for (int i = 0; i < 8; ++i) {
        int u = tid + i * 128;
        int rr = u >> 5, cc = (u & 31) * 4;
        if (n0 + rr < NN)
          *(uint2*)&sB[rr][cc] = *(const uint2*)&KV[(size_t)(b * NN + n0 + rr) * DD + cc];
        else
          *(uint2*)&sB[rr][cc] = make_uint2(0u, 0u);
      }
    }
    __syncthreads();
    v16h a;
    if (n0 + 32 <= NN) {
      a = load_frag_a(&PT[(size_t)kk0 * NN + n0], NN);
    } else {
#pragma unroll
      for (int e = 0; e < 16; ++e) {
        int kidx = (sel << 4) + e;
        int n = n0 + kidx;
        a[e] = (n < NN) ? PT[(size_t)(kk0 + m) * NN + n] : (_Float16)0.0f;
      }
    }
#pragma unroll
    for (int t = 0; t < 2; ++t) {
      int col0 = (wave * 2 + t) * 16;
      v16h bf = load_frag_b_kn(&sB[0][col0], 136);
      acc[t] = wmma_f16(a, bf, acc[t]);
    }
    __syncthreads();
  }
#pragma unroll
  for (int t = 0; t < 2; ++t) {
    int col = (wave * 2 + t) * 16 + m;
#pragma unroll
    for (int e = 0; e < 8; ++e)
      outkv[(size_t)(b * KK + kk0 + e + sel * 8) * DD + col] = (_Float16)acc[t][e];
  }
}

// ---------------------------------------------------------------------------
// Fused Linformer attention: scores(16x256) -> softmax -> P@V (16x32), per head.
// One wave32 per (b, h, 16-row tile). Probs staged via LDS to re-shape C->A layout.
// ---------------------------------------------------------------------------
__global__ void k_attn(const _Float16* __restrict__ Q, const _Float16* __restrict__ Kt,
                       const _Float16* __restrict__ V, _Float16* __restrict__ O) {
  __shared__ __align__(16) _Float16 sP[16][KK + 16];
  const int lane = threadIdx.x & 31;
  const int m = lane & 15, sel = lane >> 4;
  const int n0 = blockIdx.x * 16, h = blockIdx.y, b = blockIdx.z;
  const float scale = 0.17677669529663687f;  // DH^-0.5

  v16h aq;
#pragma unroll
  for (int e = 0; e < 16; ++e) {
    int kidx = (sel << 4) + e;
    aq[e] = Q[(size_t)(b * NN + n0 + m) * DD + h * DHH + kidx];
  }

  v8f sc[16];
#pragma unroll
  for (int t = 0; t < 16; ++t) {
    v16h bf;  // B[dh][kk] = K[b, t*16+kk, h*32+dh]
#pragma unroll
    for (int e = 0; e < 16; ++e)
      bf[e] = Kt[(size_t)(b * KK + t * 16 + m) * DD + h * DHH + (sel << 4) + e];
    v8f z = {};
    sc[t] = wmma_f16(aq, bf, z);
  }

#pragma unroll
  for (int t = 0; t < 16; ++t)
#pragma unroll
    for (int e = 0; e < 8; ++e) sc[t][e] *= scale;

  float mx[8], sm[8];
#pragma unroll
  for (int e = 0; e < 8; ++e) {
    float v = -1e30f;
#pragma unroll
    for (int t = 0; t < 16; ++t) v = fmaxf(v, sc[t][e]);
#pragma unroll
    for (int msk = 1; msk < 16; msk <<= 1) v = fmaxf(v, __shfl_xor(v, msk));
    mx[e] = v;
    sm[e] = 0.0f;
  }
#pragma unroll
  for (int t = 0; t < 16; ++t)
#pragma unroll
    for (int e = 0; e < 8; ++e) {
      float p = __expf(sc[t][e] - mx[e]);
      sc[t][e] = p;
      sm[e] += p;
    }
#pragma unroll
  for (int e = 0; e < 8; ++e) {
#pragma unroll
    for (int msk = 1; msk < 16; msk <<= 1) sm[e] += __shfl_xor(sm[e], msk);
    sm[e] = 1.0f / sm[e];
  }
#pragma unroll
  for (int t = 0; t < 16; ++t)
#pragma unroll
    for (int e = 0; e < 8; ++e)
      sP[e + 8 * sel][t * 16 + m] = (_Float16)(sc[t][e] * sm[e]);
  __syncthreads();

#pragma unroll
  for (int ct = 0; ct < 2; ++ct) {
    v8f acc = {};
#pragma unroll
    for (int s = 0; s < 8; ++s) {
      v16h a = load_frag_a(&sP[0][s * 32], KK + 16);
      v16h bf;
#pragma unroll
      for (int e = 0; e < 16; ++e)
        bf[e] = V[(size_t)(b * KK + s * 32 + (sel << 4) + e) * DD + h * DHH + ct * 16 + m];
      acc = wmma_f16(a, bf, acc);
    }
#pragma unroll
    for (int e = 0; e < 8; ++e)
      O[(size_t)(b * NN + n0 + e + 8 * sel) * DD + h * DHH + ct * 16 + m] = (_Float16)acc[e];
  }
}

// ---------------------------------------------------------------------------
// Fused FFN: mid = GELU(hn @ w1^T + b1) kept in LDS (16x512 f16), then
// xe += mid @ w2^T + b2.  8 waves per 16-row tile.
// ---------------------------------------------------------------------------
__global__ void k_ffn(const _Float16* __restrict__ Xn, const _Float16* __restrict__ W1,
                      const float* __restrict__ B1, const _Float16* __restrict__ W2,
                      const float* __restrict__ B2, float* __restrict__ XE) {
  __shared__ __align__(16) _Float16 sX[16][136];
  __shared__ __align__(16) _Float16 sM[16][4 * DD + 8];
  const int tid = threadIdx.x, wave = tid >> 5, lane = tid & 31;
  const int m = lane & 15, sel = lane >> 4;
  const int r0 = blockIdx.x * 16;

  {  // stage hn tile [16][128]
    int rr = tid >> 4, cc = (tid & 15) * 8;
#if HAS_ASYNC_LDS
    ASYNC_COPY_B128(&Xn[(size_t)(r0 + rr) * DD + cc], &sX[rr][cc]);
    ASYNC_WAIT();
#else
    *(uint4*)&sX[rr][cc] = *(const uint4*)&Xn[(size_t)(r0 + rr) * DD + cc];
#endif
  }
  __syncthreads();

  // GEMM1: [16,128] x [512,128]^T -> 32 col tiles, 4 per wave
  v8f acc[4] = {};
  for (int kt = 0; kt < 4; ++kt) {
    v16h a = load_frag_a(&sX[0][kt * 32], 136);
#pragma unroll
    for (int t = 0; t < 4; ++t) {
      int col0 = (wave * 4 + t) * 16;
      v16h bf = load_frag_b_nk(&W1[(size_t)col0 * DD + kt * 32], DD);
      acc[t] = wmma_f16(a, bf, acc[t]);
    }
  }
#pragma unroll
  for (int t = 0; t < 4; ++t) {  // bias + exact GELU -> LDS
    int col = (wave * 4 + t) * 16 + m;
    float bv = B1[col];
#pragma unroll
    for (int e = 0; e < 8; ++e) {
      float v = acc[t][e] + bv;
      float gl = 0.5f * v * (1.0f + erff(v * 0.70710678118654752f));
      sM[e + 8 * sel][col] = (_Float16)gl;
    }
  }
  __syncthreads();

  // GEMM2: [16,512] x [128,512]^T -> 8 col tiles, 1 per wave; accumulate into xe
  v8f acc2 = {};
  int col0 = wave * 16;
  for (int s = 0; s < 16; ++s) {
    v16h a = load_frag_a(&sM[0][s * 32], 4 * DD + 8);
    v16h bf = load_frag_b_nk(&W2[(size_t)col0 * (4 * DD) + s * 32], 4 * DD);
    acc2 = wmma_f16(a, bf, acc2);
  }
  int col = col0 + m;
  float bv = B2[col];
#pragma unroll
  for (int e = 0; e < 8; ++e)
    XE[(size_t)(r0 + e + 8 * sel) * DD + col] += acc2[e] + bv;
}

// ---------------------------------------------------------------------------
// Row sum over D (collapse of the final einsum's sum over d)
// ---------------------------------------------------------------------------
__global__ void k_rowsum(const float* __restrict__ xe, float* __restrict__ s) {
  int row = blockIdx.x * 8 + (threadIdx.x >> 5);
  int lane = threadIdx.x & 31;
  float4 v = *(const float4*)&xe[(size_t)row * DD + lane * 4];
  float sm = v.x + v.y + v.z + v.w;
#pragma unroll
  for (int m = 1; m < 32; m <<= 1) sm += __shfl_xor(sm, m);
  if (lane == 0) s[row] = sm;
}

// ---------------------------------------------------------------------------
// Final: out[b,m] = sum_n s[b,n] * fc_w[m,n] + D * fc_b[m]
// One wave32 per m; streams the 200 MB fc_w with float4 + prefetch.
// ---------------------------------------------------------------------------
__global__ void k_fc(const float* __restrict__ s, const float* __restrict__ fw,
                     const float* __restrict__ fb, float* __restrict__ out) {
  int mrow = blockIdx.x * 8 + (threadIdx.x >> 5);
  int lane = threadIdx.x & 31;
  float acc[BB];
#pragma unroll
  for (int b = 0; b < BB; ++b) acc[b] = 0.0f;
  const float* wr = fw + (size_t)mrow * NN;
  for (int i = 0; i < 78; ++i) {  // 78*128 = 9984
    int n = i * 128 + lane * 4;
    __builtin_prefetch(wr + n + 1024, 0, 1);
    float4 w4 = *(const float4*)&wr[n];
#pragma unroll
    for (int b = 0; b < BB; ++b) {
      float4 s4 = *(const float4*)&s[(size_t)b * NN + n];
      acc[b] += w4.x * s4.x + w4.y * s4.y + w4.z * s4.z + w4.w * s4.w;
    }
  }
  if (lane < 16) {  // tail 16
    int n = 9984 + lane;
    float w = wr[n];
#pragma unroll
    for (int b = 0; b < BB; ++b) acc[b] += w * s[(size_t)b * NN + n];
  }
#pragma unroll
  for (int b = 0; b < BB; ++b)
#pragma unroll
    for (int msk = 1; msk < 32; msk <<= 1) acc[b] += __shfl_xor(acc[b], msk);
  if (lane == 0) {
    float bias = (float)DD * fb[mrow];
#pragma unroll
    for (int b = 0; b < BB; ++b) out[(size_t)b * MM + mrow] = acc[b] + bias;
  }
}

// ---------------------------------------------------------------------------
extern "C" void kernel_launch(void* const* d_in, const int* in_sizes, int n_in,
                              void* d_out, int out_size, void* d_ws, size_t ws_size,
                              hipStream_t stream) {
  const float* x     = (const float*)d_in[0];
  const float* emb_w = (const float*)d_in[1];
  const float* emb_b = (const float*)d_in[2];
  const float* ln1_g = (const float*)d_in[3];
  const float* ln1_b = (const float*)d_in[4];
  const float* wq    = (const float*)d_in[5];
  const float* wk    = (const float*)d_in[6];
  const float* wv    = (const float*)d_in[7];
  const float* pk    = (const float*)d_in[8];
  const float* pv    = (const float*)d_in[9];
  const float* wo    = (const float*)d_in[10];
  const float* wo_b  = (const float*)d_in[11];
  const float* ln2_g = (const float*)d_in[12];
  const float* ln2_b = (const float*)d_in[13];
  const float* w1    = (const float*)d_in[14];
  const float* b1    = (const float*)d_in[15];
  const float* w2    = (const float*)d_in[16];
  const float* b2    = (const float*)d_in[17];
  const float* fcw   = (const float*)d_in[18];
  const float* fcb   = (const float*)d_in[19];
  float* out = (float*)d_out;

  char* ws = (char*)d_ws;
  size_t off = 0;
  auto alloc = [&](size_t bytes) -> char* {
    char* p = ws + off;
    off = (off + bytes + 255) & ~(size_t)255;
    return p;
  };

  float*     xe    = (float*)alloc((size_t)RTOT * DD * 4);
  _Float16*  hn16  = (_Float16*)alloc((size_t)RTOT * DD * 2);
  _Float16*  q16   = (_Float16*)alloc((size_t)RTOT * DD * 2);
  _Float16*  tmp16 = (_Float16*)alloc((size_t)RTOT * DD * 2);
  _Float16*  ao16  = (_Float16*)alloc((size_t)RTOT * DD * 2);
  _Float16*  k16   = (_Float16*)alloc((size_t)BB * KK * DD * 2);
  _Float16*  v16p  = (_Float16*)alloc((size_t)BB * KK * DD * 2);
  float*     srow  = (float*)alloc((size_t)RTOT * 4);
  _Float16*  wq16  = (_Float16*)alloc((size_t)DD * DD * 2);
  _Float16*  wk16  = (_Float16*)alloc((size_t)DD * DD * 2);
  _Float16*  wv16  = (_Float16*)alloc((size_t)DD * DD * 2);
  _Float16*  wo16  = (_Float16*)alloc((size_t)DD * DD * 2);
  _Float16*  w116  = (_Float16*)alloc((size_t)4 * DD * DD * 2);
  _Float16*  w216  = (_Float16*)alloc((size_t)4 * DD * DD * 2);
  _Float16*  pkT16 = (_Float16*)alloc((size_t)NN * KK * 2);  // [K,N]
  _Float16*  pvT16 = (_Float16*)alloc((size_t)NN * KK * 2);  // [K,N]

  auto cvt = [&](const float* src, _Float16* dst, int n) {
    k_f32_to_f16<<<(n + 255) / 256, 256, 0, stream>>>(src, dst, n);
  };
  cvt(wq, wq16, DD * DD);
  cvt(wk, wk16, DD * DD);
  cvt(wv, wv16, DD * DD);
  cvt(wo, wo16, DD * DD);
  cvt(w1, w116, 4 * DD * DD);
  cvt(w2, w216, 4 * DD * DD);
  k_cvt_transpose<<<(NN * KK + 255) / 256, 256, 0, stream>>>(pk, pkT16, NN, KK);
  k_cvt_transpose<<<(NN * KK + 255) / 256, 256, 0, stream>>>(pv, pvT16, NN, KK);

  k_embed<<<(RTOT * DD + 255) / 256, 256, 0, stream>>>(x, emb_w, emb_b, xe);
  k_layernorm<<<RTOT / 8, 256, 0, stream>>>(xe, ln1_g, ln1_b, hn16);

  k_gemm_tok<<<RTOT / 64, 256, 0, stream>>>(hn16, wq16, nullptr, q16, nullptr);
  k_gemm_tok<<<RTOT / 64, 256, 0, stream>>>(hn16, wk16, nullptr, tmp16, nullptr);
  k_proj<<<dim3(KK / 16, BB), 128, 0, stream>>>(tmp16, pkT16, k16);
  k_gemm_tok<<<RTOT / 64, 256, 0, stream>>>(hn16, wv16, nullptr, tmp16, nullptr);
  k_proj<<<dim3(KK / 16, BB), 128, 0, stream>>>(tmp16, pvT16, v16p);

  k_attn<<<dim3(NN / 16, HH, BB), 32, 0, stream>>>(q16, k16, v16p, ao16);
  k_gemm_tok<<<RTOT / 64, 256, 0, stream>>>(ao16, wo16, wo_b, nullptr, xe);

  k_layernorm<<<RTOT / 8, 256, 0, stream>>>(xe, ln2_g, ln2_b, hn16);
  k_ffn<<<RTOT / 16, 256, 0, stream>>>(hn16, w116, b1, w216, b2, xe);

  k_rowsum<<<RTOT / 8, 256, 0, stream>>>(xe, srow);
  k_fc<<<MM / 8, 256, 0, stream>>>(srow, fcw, fcb, out);
}
